// SWABlock_29875792511784
// MI455X (gfx1250) — compile-verified, matrix operand order checked by reading, not run on hardware
//
#include <hip/hip_runtime.h>

typedef __bf16 v16bf __attribute__((ext_vector_type(16)));
typedef float  v8f   __attribute__((ext_vector_type(8)));
typedef unsigned uint32x4 __attribute__((ext_vector_type(4)));
typedef unsigned uint32x8 __attribute__((ext_vector_type(8)));
typedef unsigned short u16;

#define S_LEN 2048
#define D_MOD 1024
#define D3    3072
#define NHEAD 16
#define DHEAD 64
#define WIN   128
#define MTOK  4096   // B*S

#define KC    128    // K-chunk staged in LDS per TDM transfer
#define LDA_S 136    // padded LDS row stride in bf16 elements (272B: TDM pads 4 DW / 64 DW)

// ---------------- helpers ----------------

__device__ __forceinline__ u16 f2bf(float f) {
  unsigned u = __builtin_bit_cast(unsigned, f);
  u += 0x7fffu + ((u >> 16) & 1u);          // round-to-nearest-even
  return (u16)(u >> 16);
}

__device__ __forceinline__ v8f wmma_bf16(v16bf a, v16bf b, v8f c) {
  return __builtin_amdgcn_wmma_f32_16x16x32_bf16(false, a, false, b, (short)0, c,
                                                 false, false);
}

// Per-lane fragment slice for 16-bit 16x32 A / 32x16 B operands:
// 8 contiguous bf16 at p (K = base+0..7), 8 contiguous bf16 at p+16 (K = base+16..23).
__device__ __forceinline__ v16bf load_frag(const u16* p) {
  union { v16bf v; uint4 q[2]; } f;
  f.q[0] = *reinterpret_cast<const uint4*>(p);
  f.q[1] = *reinterpret_cast<const uint4*>(p + 16);
  return f.v;
}

// ---------------- TDM: issue a 2D bf16 tile load (rows x KC) into LDS ----------------
// Tensor rows have stride k_elems (bf16 units). LDS destination gets 4 DWORDs of pad
// every 64 DWORDs (one 256B row) -> effective LDS row stride = LDA_S bf16.

__device__ __forceinline__ void tdm_issue_tile(unsigned lds_off, const u16* gtile,
                                               unsigned rows, unsigned k_elems) {
  unsigned long long ga = (unsigned long long)(size_t)gtile;
  uint32x4 g0;
  g0[0] = 1u;                                   // count=1, user descriptor
  g0[1] = lds_off;                              // lds_addr (bytes)
  g0[2] = (unsigned)ga;                         // global_addr[31:0]
  g0[3] = ((unsigned)(ga >> 32) & 0x01FFFFFFu)  // global_addr[56:32]
          | 0x80000000u;                        // type=2 ("image") in bits 127:126
  uint32x8 g1;
  g1[0] = (1u << 16)                            // data_size = 1 -> 2 bytes
        | (1u << 20)                            // pad_enable
        | (5u << 22)                            // pad_interval: 64 DWORDs
        | (3u << 25);                           // pad_amount:   4 DWORDs
  g1[1] = (k_elems & 0xFFFFu) << 16;            // tensor_dim0[15:0]
  g1[2] = (k_elems >> 16) | ((rows & 0xFFFFu) << 16);   // tensor_dim0[31:16], dim1[15:0]
  g1[3] = (rows >> 16) | ((unsigned)KC << 16);  // tensor_dim1[31:16], tile_dim0
  g1[4] = rows;                                 // tile_dim1 (tile_dim2 = 0)
  g1[5] = k_elems;                              // tensor_dim0_stride[31:0]
  g1[6] = 0u;                                   // stride0[47:32], stride1[15:0]
  g1[7] = 0u;                                   // stride1[47:16]
  asm volatile("tensor_load_to_lds %0, %1" :: "s"(g0), "s"(g1) : "memory");
}

// ---------------- f32 -> bf16 weight conversion ----------------

__global__ void cvt_bf16_kernel(const float* __restrict__ src, u16* __restrict__ dst, int n) {
  int i = blockIdx.x * 256 + threadIdx.x;
  if (i < n) dst[i] = f2bf(src[i]);
}

// ---------------- LayerNorm (D=1024), writes bf16 ----------------

__global__ __launch_bounds__(256) void ln_kernel(const float* __restrict__ x,
                                                 const float* __restrict__ w,
                                                 const float* __restrict__ b,
                                                 u16* __restrict__ out) {
  int tok = blockIdx.x;
  const float4 v = reinterpret_cast<const float4*>(x + (size_t)tok * D_MOD)[threadIdx.x];
  float s  = v.x + v.y + v.z + v.w;
  float s2 = fmaf(v.x, v.x, fmaf(v.y, v.y, fmaf(v.z, v.z, v.w * v.w)));
  #pragma unroll
  for (int o = 16; o >= 1; o >>= 1) {
    s  += __shfl_xor(s,  o, 32);
    s2 += __shfl_xor(s2, o, 32);
  }
  __shared__ float ss[8], ss2[8];
  int wv = threadIdx.x >> 5, ln = threadIdx.x & 31;
  if (ln == 0) { ss[wv] = s; ss2[wv] = s2; }
  __syncthreads();
  if (wv == 0) {
    float a  = (ln < 8) ? ss[ln]  : 0.0f;
    float a2 = (ln < 8) ? ss2[ln] : 0.0f;
    #pragma unroll
    for (int o = 4; o >= 1; o >>= 1) {
      a  += __shfl_xor(a,  o, 32);
      a2 += __shfl_xor(a2, o, 32);
    }
    if (ln == 0) { ss[0] = a; ss2[0] = a2; }
  }
  __syncthreads();
  float mean = ss[0] * (1.0f / D_MOD);
  float var  = ss2[0] * (1.0f / D_MOD) - mean * mean;
  float rstd = rsqrtf(var + 1e-5f);
  const float4 wv4 = reinterpret_cast<const float4*>(w)[threadIdx.x];
  const float4 bv4 = reinterpret_cast<const float4*>(b)[threadIdx.x];
  ushort4 o4;
  o4.x = f2bf((v.x - mean) * rstd * wv4.x + bv4.x);
  o4.y = f2bf((v.y - mean) * rstd * wv4.y + bv4.y);
  o4.z = f2bf((v.z - mean) * rstd * wv4.z + bv4.z);
  o4.w = f2bf((v.w - mean) * rstd * wv4.w + bv4.w);
  reinterpret_cast<ushort4*>(out + (size_t)tok * D_MOD)[threadIdx.x] = o4;
}

// ---------------- TDM-staged bf16 WMMA GEMM: C = act(A @ W^T + bias [+ res]) ----------------
// A: [M,K] bf16 row-major. W: [N,K] bf16 row-major.
// Block: 256 threads = 8 waves (2x4 of 32x32 wave tiles) -> 64(M) x 128(N) per block.
// K staged in KC=128 chunks via tensor_load_to_lds, double buffered, bank-conflict-free
// via TDM pad (row stride 272B).

template <bool GELU_, bool OUTBF_, bool RES_>
__global__ __launch_bounds__(256) void gemm_kernel(const u16* __restrict__ A,
                                                   const u16* __restrict__ W,
                                                   const float* __restrict__ bias,
                                                   const float* __restrict__ res,
                                                   void* __restrict__ out,
                                                   int M, int N, int K) {
  __shared__ u16 As[2][64 * LDA_S];
  __shared__ u16 Bs[2][128 * LDA_S];

  const int lane = threadIdx.x & 31;
  const int wave = threadIdx.x >> 5;
  const int hl   = lane >> 4;
  const int l16  = lane & 15;
  const int wml = (wave >> 2) * 32;            // local row within block tile
  const int wnl = (wave & 3) * 32;             // local col within block tile
  const int wm = blockIdx.y * 64  + wml;
  const int wn = blockIdx.x * 128 + wnl;

  const u16* Ag = A + (size_t)blockIdx.y * 64  * K;   // block's A tile origin
  const u16* Bg = W + (size_t)blockIdx.x * 128 * K;   // block's B tile origin
  const unsigned ldsA = (unsigned)(size_t)&As[0][0];
  const unsigned ldsB = (unsigned)(size_t)&Bs[0][0];
  const unsigned stA = 64 * LDA_S * 2;                // stage byte sizes
  const unsigned stB = 128 * LDA_S * 2;

  const bool issuer = (wave == 0);
  if (issuer) {
    tdm_issue_tile(ldsA, Ag, 64u,  (unsigned)K);
    tdm_issue_tile(ldsB, Bg, 128u, (unsigned)K);
  }

  v8f acc00 = {}, acc01 = {}, acc10 = {}, acc11 = {};

  int st = 0;
  for (int k0 = 0; k0 < K; k0 += KC, st ^= 1) {
    if (issuer) {
      if (k0 + KC < K) {
        tdm_issue_tile(ldsA + (unsigned)(st ^ 1) * stA, Ag + (k0 + KC), 64u,  (unsigned)K);
        tdm_issue_tile(ldsB + (unsigned)(st ^ 1) * stB, Bg + (k0 + KC), 128u, (unsigned)K);
        __builtin_amdgcn_s_wait_tensorcnt(2);   // current stage's 2 transfers done
      } else {
        __builtin_amdgcn_s_wait_tensorcnt(0);
      }
    }
    __syncthreads();                            // stage st ready for all waves

    const u16* a0p = &As[st][(wml + l16) * LDA_S + hl * 8];
    const u16* b0p = &Bs[st][(wnl + l16) * LDA_S + hl * 8];
    #pragma unroll
    for (int kk = 0; kk < KC; kk += 32) {
      v16bf a0 = load_frag(a0p + kk);
      v16bf a1 = load_frag(a0p + 16 * LDA_S + kk);
      v16bf b0 = load_frag(b0p + kk);
      v16bf b1 = load_frag(b0p + 16 * LDA_S + kk);
      acc00 = wmma_bf16(a0, b0, acc00);
      acc01 = wmma_bf16(a0, b1, acc01);
      acc10 = wmma_bf16(a1, b0, acc10);
      acc11 = wmma_bf16(a1, b1, acc11);
    }
    __syncthreads();                            // stage st free for next TDM write
  }

  const int r8 = hl * 8;
  #pragma unroll
  for (int r = 0; r < 8; r++) {
    int row0 = wm + r + r8, row1 = row0 + 16;
    int col0 = wn + l16,    col1 = col0 + 16;
    float vals[4] = { acc00[r], acc01[r], acc10[r], acc11[r] };
    int rows[4] = { row0, row0, row1, row1 };
    int cols[4] = { col0, col1, col0, col1 };
    #pragma unroll
    for (int t = 0; t < 4; t++) {
      float v = vals[t] + bias[cols[t]];
      size_t idx = (size_t)rows[t] * N + cols[t];
      if (RES_)  v += res[idx];
      if (GELU_) v = 0.5f * v * (1.0f + erff(v * 0.70710678118654752f));
      if (OUTBF_) ((u16*)out)[idx] = f2bf(v);
      else        ((float*)out)[idx] = v;
    }
  }
}

// ---------------- repack qkv -> Q [B,H,S,DH], K [B,H,S,DH], Vt [B,H,DH,S] ----------------

__global__ void repack_qkv_kernel(const u16* __restrict__ qkv, u16* __restrict__ Q,
                                  u16* __restrict__ Kt, u16* __restrict__ Vt) {
  size_t idx = (size_t)blockIdx.x * 256 + threadIdx.x;  // over B*H*S*DH = 4M
  int d  = idx & 63;
  int s  = (int)((idx >> 6) & (S_LEN - 1));
  int bh = (int)(idx >> 17);
  int b  = bh >> 4, h = bh & 15;
  size_t src = ((size_t)(b * S_LEN + s)) * D3 + h * DHEAD + d;
  Q[idx]  = qkv[src];
  Kt[idx] = qkv[src + D_MOD];
  Vt[((size_t)bh * DHEAD + d) * S_LEN + s] = qkv[src + 2 * D_MOD];
}

// ---------------- sliding-window attention (flash style, 1 wave / 16 queries) ----------------

__global__ __launch_bounds__(32) void attn_kernel(const u16* __restrict__ Q,
                                                  const u16* __restrict__ Kt,
                                                  const u16* __restrict__ Vt,
                                                  u16* __restrict__ ctx) {
  const int lane = threadIdx.x;
  const int hl = lane >> 4, l16 = lane & 15;
  const int bh = blockIdx.y;
  const int qi = blockIdx.x * 16;

  const u16* Qb = Q  + ((size_t)bh * S_LEN + qi) * DHEAD;
  const u16* Kb = Kt + (size_t)bh * S_LEN * DHEAD;
  const u16* Vb = Vt + (size_t)bh * DHEAD * S_LEN;

  const u16* qp = Qb + (size_t)l16 * DHEAD + hl * 8;
  v16bf aq0 = load_frag(qp);
  v16bf aq1 = load_frag(qp + 32);

  v8f o0 = {}, o1 = {}, o2 = {}, o3 = {};
  float mrow[8], lrow[8];
  #pragma unroll
  for (int r = 0; r < 8; r++) { mrow[r] = -1e30f; lrow[r] = 0.0f; }

  __shared__ u16 Pl[16 * 32];

  int c_lo = qi - WIN + 1;
  if (c_lo < 0) c_lo = 0;
  c_lo &= ~31;

  for (int c = c_lo; c <= qi + 15; c += 32) {
    const u16* kp0 = Kb + (size_t)(c + l16) * DHEAD + hl * 8;
    const u16* kp1 = kp0 + (size_t)16 * DHEAD;
    v8f s0 = {}, s1 = {};
    s0 = wmma_bf16(aq0, load_frag(kp0), s0);
    s0 = wmma_bf16(aq1, load_frag(kp0 + 32), s0);
    s1 = wmma_bf16(aq0, load_frag(kp1), s1);
    s1 = wmma_bf16(aq1, load_frag(kp1 + 32), s1);

    float alpha[8], p0[8], p1[8];
    #pragma unroll
    for (int r = 0; r < 8; r++) {
      int row = qi + r + hl * 8;
      int j0 = c + l16, j1 = j0 + 16;
      bool a0 = (j0 <= row) && (j0 > row - WIN);
      bool a1 = (j1 <= row) && (j1 > row - WIN);
      float v0 = s0[r] * 0.125f;   // 1/sqrt(64)
      float v1 = s1[r] * 0.125f;
      float mv = fmaxf(a0 ? v0 : -1e30f, a1 ? v1 : -1e30f);
      #pragma unroll
      for (int o = 8; o >= 1; o >>= 1) mv = fmaxf(mv, __shfl_xor(mv, o, 32));
      float mn = fmaxf(mrow[r], mv);
      float al = __expf(mrow[r] - mn);
      p0[r] = a0 ? __expf(v0 - mn) : 0.0f;
      p1[r] = a1 ? __expf(v1 - mn) : 0.0f;
      float ps = p0[r] + p1[r];
      #pragma unroll
      for (int o = 8; o >= 1; o >>= 1) ps += __shfl_xor(ps, o, 32);
      lrow[r] = lrow[r] * al + ps;
      mrow[r] = mn;
      alpha[r] = al;
    }
    #pragma unroll
    for (int r = 0; r < 8; r++) {
      o0[r] *= alpha[r]; o1[r] *= alpha[r]; o2[r] *= alpha[r]; o3[r] *= alpha[r];
    }

    // P (16x32, C layout) -> LDS row-major bf16 -> reload as A-fragment
    #pragma unroll
    for (int r = 0; r < 8; r++) {
      int prow = r + hl * 8;
      Pl[prow * 32 + l16]      = f2bf(p0[r]);
      Pl[prow * 32 + l16 + 16] = f2bf(p1[r]);
    }
    asm volatile("s_wait_dscnt 0x0" ::: "memory");
    v16bf pa = load_frag(&Pl[l16 * 32 + hl * 8]);

    const size_t cofs = (size_t)c + hl * 8;
    o0 = wmma_bf16(pa, load_frag(Vb + (size_t)(0  + l16) * S_LEN + cofs), o0);
    o1 = wmma_bf16(pa, load_frag(Vb + (size_t)(16 + l16) * S_LEN + cofs), o1);
    o2 = wmma_bf16(pa, load_frag(Vb + (size_t)(32 + l16) * S_LEN + cofs), o2);
    o3 = wmma_bf16(pa, load_frag(Vb + (size_t)(48 + l16) * S_LEN + cofs), o3);
  }

  const int b = bh >> 4, h = bh & 15;
  #pragma unroll
  for (int r = 0; r < 8; r++) {
    float inv = 1.0f / lrow[r];
    int row = qi + r + hl * 8;
    size_t obase = ((size_t)(b * S_LEN + row)) * D_MOD + h * DHEAD + l16;
    ctx[obase +  0] = f2bf(o0[r] * inv);
    ctx[obase + 16] = f2bf(o1[r] * inv);
    ctx[obase + 32] = f2bf(o2[r] * inv);
    ctx[obase + 48] = f2bf(o3[r] * inv);
  }
}

// ---------------- launch ----------------

extern "C" void kernel_launch(void* const* d_in, const int* in_sizes, int n_in,
                              void* d_out, int out_size, void* d_ws, size_t ws_size,
                              hipStream_t stream) {
  const float* x     = (const float*)d_in[0];
  const float* qkv_w = (const float*)d_in[1];
  const float* qkv_b = (const float*)d_in[2];
  const float* out_w = (const float*)d_in[3];
  const float* out_b = (const float*)d_in[4];
  const float* ff1_w = (const float*)d_in[5];
  const float* ff1_b = (const float*)d_in[6];
  const float* ff2_w = (const float*)d_in[7];
  const float* ff2_b = (const float*)d_in[8];
  const float* ln1_w = (const float*)d_in[9];
  const float* ln1_b = (const float*)d_in[10];
  const float* ln2_w = (const float*)d_in[11];
  const float* ln2_b = (const float*)d_in[12];
  float* out = (float*)d_out;

  char* ws = (char*)d_ws;
  size_t off = 0;
  auto alloc = [&](size_t bytes) {
    void* p = ws + off;
    off += (bytes + 255) & ~(size_t)255;
    return p;
  };
  const size_t M = MTOK;
  u16* h1   = (u16*)alloc(M * D_MOD * 2);          // LN1 out, bf16
  u16* qkva = (u16*)alloc(M * D3 * 2);             // qkv activations, bf16
  u16* Qb   = (u16*)alloc(M * D_MOD * 2);          // [B,H,S,DH]
  u16* Kb   = (u16*)alloc(M * D_MOD * 2);          // [B,H,S,DH]
  u16* Vt   = (u16*)alloc(M * D_MOD * 2);          // [B,H,DH,S]
  u16* ctx  = (u16*)alloc(M * D_MOD * 2);          // attention output, bf16
  float* x1 = (float*)alloc(M * D_MOD * 4);        // after out-proj residual, f32
  u16* h2   = (u16*)alloc(M * D_MOD * 2);          // LN2 out, bf16
  u16* ffa  = (u16*)alloc(M * D3 * 2);             // GELU(FF1) activations, bf16
  u16* wqkv = (u16*)alloc((size_t)D3 * D_MOD * 2);
  u16* wout = (u16*)alloc((size_t)D_MOD * D_MOD * 2);
  u16* wff1 = (u16*)alloc((size_t)D3 * D_MOD * 2);
  u16* wff2 = (u16*)alloc((size_t)D_MOD * D3 * 2);

  // 1) weight conversion to bf16 (layout preserved: [N,K])
  {
    int n1 = D3 * D_MOD, n2 = D_MOD * D_MOD;
    cvt_bf16_kernel<<<(n1 + 255) / 256, 256, 0, stream>>>(qkv_w, wqkv, n1);
    cvt_bf16_kernel<<<(n2 + 255) / 256, 256, 0, stream>>>(out_w, wout, n2);
    cvt_bf16_kernel<<<(n1 + 255) / 256, 256, 0, stream>>>(ff1_w, wff1, n1);
    cvt_bf16_kernel<<<(n1 + 255) / 256, 256, 0, stream>>>(ff2_w, wff2, n1);
  }

  // 2) LN1
  ln_kernel<<<M, 256, 0, stream>>>(x, ln1_w, ln1_b, h1);

  // 3) QKV projection: [4096,1024] x [3072,1024]^T -> bf16 [4096,3072]
  gemm_kernel<false, true, false>
      <<<dim3(D3 / 128, M / 64), 256, 0, stream>>>(h1, wqkv, qkv_b, nullptr, qkva,
                                                   (int)M, D3, D_MOD);

  // 4) repack into Q/K [B,H,S,DH] and V transposed [B,H,DH,S]
  repack_qkv_kernel<<<(M * D_MOD) / 256, 256, 0, stream>>>(qkva, Qb, Kb, Vt);

  // 5) sliding-window attention
  attn_kernel<<<dim3(S_LEN / 16, 2 * NHEAD), 32, 0, stream>>>(Qb, Kb, Vt, ctx);

  // 6) out projection + residual(x): f32 [4096,1024]
  gemm_kernel<false, false, true>
      <<<dim3(D_MOD / 128, M / 64), 256, 0, stream>>>(ctx, wout, out_b, x, x1,
                                                      (int)M, D_MOD, D_MOD);

  // 7) LN2
  ln_kernel<<<M, 256, 0, stream>>>(x1, ln2_w, ln2_b, h2);

  // 8) FF1 + exact GELU -> bf16 [4096,3072]
  gemm_kernel<true, true, false>
      <<<dim3(D3 / 128, M / 64), 256, 0, stream>>>(h2, wff1, ff1_b, nullptr, ffa,
                                                   (int)M, D3, D_MOD);

  // 9) FF2 + residual(x1) -> final f32 output
  gemm_kernel<false, false, true>
      <<<dim3(D_MOD / 128, M / 64), 256, 0, stream>>>(ffa, wff2, ff2_b, x1, out,
                                                      (int)M, D_MOD, D3);
}
